// RNN_28209345200270
// MI455X (gfx1250) — matrix operand order repaired and checked by reading, not compiled, not run on hardware
//
#include <hip/hip_runtime.h>
#include <math.h>

// Problem constants (match reference)
constexpr int SEQ = 512;   // seq len
constexpr int NB  = 64;    // batch
constexpr int NI  = 512;   // input size
constexpr int NH  = 1024;  // hidden size

typedef __attribute__((ext_vector_type(16))) __bf16 v16bf;
typedef __attribute__((ext_vector_type(8)))  __bf16 v8bf;
typedef __attribute__((ext_vector_type(8)))  float  v8f;

union FragBF { v16bf v; v8bf h[2]; };

// ---------------------------------------------------------------------------
// Fragment loaders per CDNA5 WMMA VGPR layouts (CDNA5 ISA §7.12.2)
// A (16-bit, 16x32 MxK): lane holds row M = lane%16; elements are two
// contiguous 8-wide K chunks: K = hv*8+{0..7} and K = 16+hv*8+{0..7},
// hv = lane/16.
// ---------------------------------------------------------------------------
__device__ __forceinline__ v16bf load_a_bf16(const __bf16* __restrict__ base,
                                             int ld, int m0, int k0, int lane) {
  const int m  = m0 + (lane & 15);
  const int hv = (lane >> 4) << 3;           // 0 or 8
  const __bf16* p = base + (size_t)m * ld + k0 + hv;
  FragBF u;
  u.h[0] = *reinterpret_cast<const v8bf*>(p);
  u.h[1] = *reinterpret_cast<const v8bf*>(p + 16);
  return u.v;
}

// Same A layout, but source is f32: convert in-register to bf16.
__device__ __forceinline__ v16bf load_a_f32(const float* __restrict__ base,
                                            int ld, int m0, int k0, int lane) {
  const int m  = m0 + (lane & 15);
  const int hv = (lane >> 4) << 3;
  const float* p = base + (size_t)m * ld + k0 + hv;
  FragBF u;
#pragma unroll
  for (int j = 0; j < 8; ++j) u.h[0][j] = (__bf16)p[j];
#pragma unroll
  for (int j = 0; j < 8; ++j) u.h[1][j] = (__bf16)p[16 + j];
  return u.v;
}

// B (16-bit, 32x16 KxN): lane holds column N = lane%16; elements are
// K = hv*8+{0..7} and K = 16+hv*8+{0..7}, hv = lane/16 (strided loads).
__device__ __forceinline__ v16bf load_b_bf16(const __bf16* __restrict__ base,
                                             int ld, int k0, int n0, int lane) {
  const int hv = (lane >> 4) << 3;
  const __bf16* p = base + (size_t)(k0 + hv) * ld + n0 + (lane & 15);
  FragBF u;
#pragma unroll
  for (int j = 0; j < 8; ++j) u.h[0][j] = p[(size_t)j * ld];
#pragma unroll
  for (int j = 0; j < 8; ++j) u.h[1][j] = p[(size_t)(16 + j) * ld];
  return u.v;
}

__device__ __forceinline__ v8f wmma_bf16(v16bf a, v16bf b, v8f c) {
  return __builtin_amdgcn_wmma_f32_16x16x32_bf16(false, a, false, b,
                                                 (short)0, c, false, false);
}

// ---------------------------------------------------------------------------
// prep: Wt[k*NH+n] = bf16(Ww[n*NH+k]); Vt[i*NH+h] = bf16(Vw[h*NI+i]);
//       bias[h] = Wb[h] + Vb[h]
// ---------------------------------------------------------------------------
__global__ void prep_kernel(const float* __restrict__ Ww, const float* __restrict__ Vw,
                            const float* __restrict__ Wb, const float* __restrict__ Vb,
                            __bf16* __restrict__ Wt, __bf16* __restrict__ Vt,
                            float* __restrict__ bias) {
  const int idx = blockIdx.x * blockDim.x + threadIdx.x;
  if (idx < NH * NH) {
    const int k = idx / NH, n = idx % NH;
    Wt[idx] = (__bf16)Ww[(size_t)n * NH + k];
  }
  if (idx < NI * NH) {
    const int k = idx / NH, n = idx % NH;
    Vt[idx] = (__bf16)Vw[(size_t)n * NI + k];
  }
  if (idx < NH) bias[idx] = Wb[idx] + Vb[idx];
}

__global__ void zero_h_kernel(__bf16* __restrict__ h) {
  const int idx = blockIdx.x * blockDim.x + threadIdx.x;
  if (idx < NB * NH) h[idx] = (__bf16)0.0f;
}

// ---------------------------------------------------------------------------
// vx: out[(s*NB+b)*NH + h] = sum_i x[s,b,i] * Vt[i][h] + bias[h]
// M = SEQ*NB = 32768, K = NI = 512, N = NH = 1024.
// Tile: 128(M) x 64(N) per WG; 8 waves, each 32x32 (2x2 WMMA tiles).
// x and Vt are L2-resident (64 MB + 1 MB << 192 MB) -> direct global loads.
// ---------------------------------------------------------------------------
__global__ void __launch_bounds__(256)
vx_kernel(const float* __restrict__ x, const __bf16* __restrict__ Vt,
          const float* __restrict__ bias, float* __restrict__ out) {
  const int wave = threadIdx.x >> 5;
  const int lane = threadIdx.x & 31;
  const int m0 = blockIdx.x * 128 + (wave & 3) * 32;
  const int n0 = blockIdx.y * 64  + (wave >> 2) * 32;

  v8f acc[2][2] = {};
#pragma unroll 4
  for (int k0 = 0; k0 < NI; k0 += 32) {
    v16bf a0 = load_a_f32(x, NI, m0,      k0, lane);
    v16bf a1 = load_a_f32(x, NI, m0 + 16, k0, lane);
    v16bf b0 = load_b_bf16(Vt, NH, k0, n0,      lane);
    v16bf b1 = load_b_bf16(Vt, NH, k0, n0 + 16, lane);
    acc[0][0] = wmma_bf16(a0, b0, acc[0][0]);
    acc[0][1] = wmma_bf16(a0, b1, acc[0][1]);
    acc[1][0] = wmma_bf16(a1, b0, acc[1][0]);
    acc[1][1] = wmma_bf16(a1, b1, acc[1][1]);
  }

  const int hvm = 8 * (lane >> 4);
#pragma unroll
  for (int nt = 0; nt < 2; ++nt) {
    const int n = n0 + nt * 16 + (lane & 15);
    const float bn = bias[n];
#pragma unroll
    for (int mt = 0; mt < 2; ++mt) {
#pragma unroll
      for (int p = 0; p < 8; ++p) {
        const int m = m0 + mt * 16 + p + hvm;
        out[(size_t)m * NH + n] = acc[mt][nt][p] + bn;
      }
    }
  }
}

// ---------------------------------------------------------------------------
// One recurrence step: h = tanh(hprev @ Wt + io), io holds Vx_t + bias (f32).
// Grid: NH/16 = 64 WGs (one 16-col slab each). 8 waves split K=1024 into
// 128-wide slices; partial 64x16 f32 tiles reduced through LDS.
// Writes h back over io (f32) and a bf16 copy for the next step's A operand.
// ---------------------------------------------------------------------------
__global__ void __launch_bounds__(256)
step_kernel(const __bf16* __restrict__ hprev, const __bf16* __restrict__ Wt,
            float* __restrict__ io, __bf16* __restrict__ hnext,
            float* __restrict__ fin) {
  const int n0   = blockIdx.x * 16;
  const int wave = threadIdx.x >> 5;
  const int lane = threadIdx.x & 31;
  const int kb   = wave * 128;

  v8f acc[4] = {};
#pragma unroll
  for (int kc = 0; kc < 128; kc += 32) {
    const int k0 = kb + kc;
    // prefetch next K-chunk of the weight slab (global_prefetch_b8)
    if (kc + 32 < 128)
      __builtin_prefetch(Wt + (size_t)(k0 + 32 + lane) * NH + n0, 0, 1);
    v16bf bf = load_b_bf16(Wt, NH, k0, n0, lane);
#pragma unroll
    for (int mt = 0; mt < 4; ++mt) {
      v16bf af = load_a_bf16(hprev, NH, mt * 16, k0, lane);
      acc[mt] = wmma_bf16(af, bf, acc[mt]);
    }
  }

  __shared__ float red[8][NB * 16];  // 32 KB (of 320 KB/WGP)
  const int hvm = 8 * (lane >> 4);
#pragma unroll
  for (int mt = 0; mt < 4; ++mt)
#pragma unroll
    for (int p = 0; p < 8; ++p)
      red[wave][(mt * 16 + p + hvm) * 16 + (lane & 15)] = acc[mt][p];
  __syncthreads();

  for (int i = threadIdx.x; i < NB * 16; i += 256) {
    float s = 0.0f;
#pragma unroll
    for (int w = 0; w < 8; ++w) s += red[w][i];
    const int m = i >> 4;
    const int n = n0 + (i & 15);
    const float h = tanhf(s + io[(size_t)m * NH + n]);
    io[(size_t)m * NH + n]    = h;
    hnext[(size_t)m * NH + n] = (__bf16)h;
    if (fin) fin[(size_t)m * NH + n] = h;
  }
}

// ---------------------------------------------------------------------------
extern "C" void kernel_launch(void* const* d_in, const int* in_sizes, int n_in,
                              void* d_out, int out_size, void* d_ws, size_t ws_size,
                              hipStream_t stream) {
  (void)in_sizes; (void)n_in; (void)out_size; (void)ws_size;
  const float* x  = (const float*)d_in[0];  // (S,B,I)
  const float* Ww = (const float*)d_in[1];  // (H,H)
  const float* Wb = (const float*)d_in[2];  // (H)
  const float* Vw = (const float*)d_in[3];  // (H,I)
  const float* Vb = (const float*)d_in[4];  // (H)
  float* out = (float*)d_out;               // S*B*H h_seq  ++  B*H h_final

  // workspace carve-up (~3.3 MB)
  char* ws = (char*)d_ws;
  __bf16* Wt    = (__bf16*)ws;                               // H*H bf16 (2 MB)
  __bf16* Vt    = (__bf16*)(ws + (size_t)NH * NH * 2);       // I*H bf16 (1 MB)
  float*  bias  = (float*) (ws + (size_t)NH * NH * 2 + (size_t)NI * NH * 2);
  __bf16* hbuf0 = (__bf16*)((char*)bias + (size_t)NH * 4);   // B*H bf16
  __bf16* hbuf1 = hbuf0 + (size_t)NB * NH;                   // B*H bf16

  // 1) transpose/convert weights, fuse biases
  prep_kernel<<<(NH * NH + 255) / 256, 256, 0, stream>>>(Ww, Vw, Wb, Vb, Wt, Vt, bias);
  // 2) h_0 = 0 (re-done every call: deterministic)
  zero_h_kernel<<<(NB * NH + 255) / 256, 256, 0, stream>>>(hbuf0);
  // 3) Vx + bias -> d_out (in-place input to the scan)
  vx_kernel<<<dim3((SEQ * NB) / 128, NH / 64), 256, 0, stream>>>(x, Vt, bias, out);
  // 4) sequential scan; ping-pong bf16 hidden state
  for (int s = 0; s < SEQ; ++s) {
    const __bf16* hp = (s & 1) ? hbuf1 : hbuf0;
    __bf16*       hn = (s & 1) ? hbuf0 : hbuf1;
    float* fin = (s == SEQ - 1) ? (out + (size_t)SEQ * NB * NH) : nullptr;
    step_kernel<<<NH / 16, 256, 0, stream>>>(hp, Wt, out + (size_t)s * NB * NH, hn, fin);
  }
}
